// GatedLayer_46359876993420
// MI455X (gfx1250) — matrix operand order repaired and checked by reading, not compiled, not run on hardware
//
#include <hip/hip_runtime.h>
#include <hip/hip_bf16.h>
#include <stddef.h>

// Problem constants (match reference)
#define Bsz   4096
#define Kdim  1024   // D_IN
#define Ndim  1024   // D_OUT
#define Mexp  16

// GEMM tiling
#define BM 128
#define BN 128
#define BK 32
#define NK (Kdim / BK)
#define LDA 40        // padded LDS row stride in bf16 (80 bytes, 16B-aligned, conflict-spreading)
#define BUFELTS (BM * LDA)
#define BUFBYTES ((unsigned)(BUFELTS * 2))    // 10240 bytes per LDS buffer

typedef __attribute__((ext_vector_type(16))) __bf16 v16bf;
typedef __attribute__((ext_vector_type(8)))  __bf16 v8bf;
typedef __attribute__((ext_vector_type(4)))  __bf16 v4bf;
typedef __attribute__((ext_vector_type(8)))  float  v8f;

union ABFrag { v16bf v; v8bf h[2]; };

// Async DMA: global (SADDR base + 32-bit VGPR offset) -> LDS, 16B per lane.
// Tracked by ASYNCcnt; no VGPR data registers involved.
#define ASYNC_LOAD_B128(ldsOff, vOff, sBase)                                   \
    asm volatile("global_load_async_to_lds_b128 %0, %1, %2 offset:0"           \
                 :: "v"(ldsOff), "v"(vOff), "s"(sBase) : "memory")

// ---------------------------------------------------------------------------
// 1) x fp32 -> bf16, 4 elements per thread
// ---------------------------------------------------------------------------
__global__ __launch_bounds__(256) void cvt_x_kernel(const float* __restrict__ x,
                                                    __bf16* __restrict__ xb) {
    size_t i = ((size_t)blockIdx.x * 256 + threadIdx.x) * 4;
    float4 v = *(const float4*)(x + i);
    v4bf o;
    o.x = (__bf16)v.x; o.y = (__bf16)v.y; o.z = (__bf16)v.z; o.w = (__bf16)v.w;
    *(v4bf*)(xb + i) = o;
}

// ---------------------------------------------------------------------------
// 2) W[m][k][n] fp32 -> Wt[m][n][k] bf16 (LDS-tiled 32x32 transpose)
//    grid.x = Mexp * (Kdim/32) * (Ndim/32)
// ---------------------------------------------------------------------------
__global__ __launch_bounds__(256) void cvt_wT_kernel(const float* __restrict__ W,
                                                     __bf16* __restrict__ Wt) {
    __shared__ float tile[32][33];
    int bid = blockIdx.x;
    int m   = bid >> 10;            // / (32*32)
    int rem = bid & 1023;
    int kt  = rem >> 5;             // k-tile
    int nt  = rem & 31;             // n-tile

    int t   = threadIdx.x;
    int col = t & 31;               // fast index
    int rr  = t >> 5;               // 0..7

    const float* src = W + ((size_t)m * Kdim + (size_t)kt * 32) * Ndim + (size_t)nt * 32;
#pragma unroll
    for (int i = 0; i < 4; ++i) {
        int r = rr + i * 8;
        tile[r][col] = src[(size_t)r * Ndim + col];
    }
    __syncthreads();
    __bf16* dst = Wt + ((size_t)m * Ndim + (size_t)nt * 32) * Kdim + (size_t)kt * 32;
#pragma unroll
    for (int i = 0; i < 4; ++i) {
        int r = rr + i * 8;                         // n within tile
        dst[(size_t)r * Kdim + col] = (__bf16)tile[col][r];
    }
}

// ---------------------------------------------------------------------------
// 3) gates[b][m] = relu( (x@Wg + bg)^2 / rowsum ) — one wave32 per row
//    grid.x = Bsz/8, block = 256 (8 waves)
// ---------------------------------------------------------------------------
__global__ __launch_bounds__(256) void gate_kernel(const float* __restrict__ x,
                                                   const float* __restrict__ Wg,
                                                   const float* __restrict__ bg,
                                                   float* __restrict__ gates) {
    int lane = threadIdx.x & 31;
    int wv   = threadIdx.x >> 5;
    int row  = blockIdx.x * 8 + wv;

    const float* xr = x + (size_t)row * Kdim;
    float acc[Mexp];
#pragma unroll
    for (int m = 0; m < Mexp; ++m) acc[m] = 0.0f;

    for (int k = lane; k < Kdim; k += 32) {
        float xv = xr[k];
        const float* wgk = Wg + (size_t)k * Mexp;
#pragma unroll
        for (int m = 0; m < Mexp; ++m) acc[m] = fmaf(xv, wgk[m], acc[m]);
    }
#pragma unroll
    for (int m = 0; m < Mexp; ++m) {
#pragma unroll
        for (int off = 16; off > 0; off >>= 1)
            acc[m] += __shfl_xor(acc[m], off, 32);
    }
    float gp[Mexp];
    float total = 0.0f;
#pragma unroll
    for (int m = 0; m < Mexp; ++m) {
        float gu = acc[m] + bg[m];
        gp[m] = gu * gu;
        total += gp[m];
    }
    if (lane < Mexp)
        gates[(size_t)row * Mexp + lane] = fmaxf(gp[lane] / total, 0.0f);
}

// ---------------------------------------------------------------------------
// 4) Fused MoE GEMM: out[b,n] = sum_m gates[b,m]*relu((x@W_m)[b,n] + b_m[n])
//    grid = (Bsz/BM, Ndim/BN), block = 256 (8 waves: 2 in M x 4 in N)
//    Each wave: 64(M) x 32(N) via 4x2 v_wmma_f32_16x16x32_bf16 tiles.
//    Depth-2 software pipeline: 3 LDS buffers filled by
//    GLOBAL_LOAD_ASYNC_TO_LDS_B128 (ASYNCcnt), 1 barrier / K-step.
//    Last two K-steps are peeled so the steady-state inner loop is a single
//    straight-line block (no wait branches, no %3 arithmetic).
// ---------------------------------------------------------------------------
__global__ __launch_bounds__(256) void moe_wmma_kernel(const __bf16* __restrict__ xb,
                                                       const __bf16* __restrict__ wt,
                                                       const float*  __restrict__ bias,
                                                       const float*  __restrict__ gates,
                                                       float* __restrict__ out) {
    __shared__ __align__(16) __bf16 As[3][BUFELTS];
    __shared__ __align__(16) __bf16 Bs[3][BUFELTS];
    __shared__ __align__(32) float  Gs[BM];

    const int t      = threadIdx.x;
    const int lane   = t & 31;
    const int wave   = t >> 5;
    const int wm     = wave & 1;   // *64 in M
    const int wn     = wave >> 1;  // *32 in N
    const int laneLo = lane & 15;
    const int half   = lane >> 4;

    const int bm0 = blockIdx.x * BM;
    const int bn0 = blockIdx.y * BN;

    // Per-thread staging map: 128 rows * 4 chunks of 8 bf16 = 512 chunks,
    // 256 threads -> 2 chunks each (same map for A and B tiles).
    // loff: LDS byte offset within a buffer; voA/voB: global byte offsets
    // (K-advance is folded into the uniform SADDR base instead).
    unsigned loff[2], voA[2], voB[2];
#pragma unroll
    for (int i = 0; i < 2; ++i) {
        int idx = t + i * 256;
        int row = idx >> 2;
        int c   = idx & 3;
        loff[i] = (unsigned)((row * LDA + c * 8) * 2);
        voA[i]  = (unsigned)(((bm0 + row) * Kdim + c * 8) * 2);
        voB[i]  = (unsigned)(((bn0 + row) * Kdim + c * 8) * 2);
    }
    const unsigned aBase0 = (unsigned)(size_t)&As[0][0];
    const unsigned bBase0 = (unsigned)(size_t)&Bs[0][0];
    const unsigned long long xbase = (unsigned long long)(size_t)xb;

    int col[2];
#pragma unroll
    for (int nt = 0; nt < 2; ++nt) col[nt] = bn0 + wn * 32 + nt * 16 + laneLo;

    // Fragment LDS element offsets (relative to a buffer base)
    int aOff[4], bOff[2];
#pragma unroll
    for (int mt = 0; mt < 4; ++mt)
        aOff[mt] = (wm * 64 + mt * 16 + laneLo) * LDA + half * 8;
#pragma unroll
    for (int nt = 0; nt < 2; ++nt)
        bOff[nt] = (wn * 32 + nt * 16 + laneLo) * LDA + half * 8;

    v8f outAcc[4][2];
#pragma unroll
    for (int mt = 0; mt < 4; ++mt)
#pragma unroll
        for (int nt = 0; nt < 2; ++nt)
            outAcc[mt][nt] = (v8f){0, 0, 0, 0, 0, 0, 0, 0};

#pragma unroll 1
    for (int m = 0; m < Mexp; ++m) {
        const unsigned long long wbase =
            (unsigned long long)(size_t)(wt + (size_t)m * Ndim * Kdim);

        __syncthreads();  // prior epilogue Gs reads + prior k-loop LDS reads done
        if (t < BM) Gs[t] = gates[(size_t)(bm0 + t) * Mexp + m];

        float bv[2];
#pragma unroll
        for (int nt = 0; nt < 2; ++nt) bv[nt] = bias[(size_t)m * Ndim + col[nt]];

        v8f acc[4][2];
#pragma unroll
        for (int mt = 0; mt < 4; ++mt)
#pragma unroll
            for (int nt = 0; nt < 2; ++nt)
                acc[mt][nt] = (v8f){0, 0, 0, 0, 0, 0, 0, 0};

        // ---- helpers (inlined) ------------------------------------------
        auto issue_tile = [&](int ktile, unsigned aBuf, unsigned bBuf) {
            const unsigned long long sA = xbase + (unsigned)(ktile * BK * 2);
            const unsigned long long sB = wbase + (unsigned)(ktile * BK * 2);
#pragma unroll
            for (int i = 0; i < 2; ++i) {
                ASYNC_LOAD_B128(aBuf + loff[i], voA[i], sA);
                ASYNC_LOAD_B128(bBuf + loff[i], voB[i], sB);
            }
        };
        auto frag_wmma = [&](const __bf16* aC, const __bf16* bC) {
            // A frags: 16-bit 16x32 layout — lanes 0-15: K 0..7 & 16..23,
            // lanes 16-31: K 8..15 & 24..31, row M = laneLo.
            ABFrag afr[4];
#pragma unroll
            for (int mt = 0; mt < 4; ++mt) {
                afr[mt].h[0] = *(const v8bf*)(aC + aOff[mt]);
                afr[mt].h[1] = *(const v8bf*)(aC + aOff[mt] + 16);
            }
            // B frags: 32x16 layout — lanes 0-15: N=laneLo, K 0..7 & 16..23;
            // lanes 16-31: N=laneLo, K 8..15 & 24..31 (K-contiguous thanks to W^T).
            ABFrag bfr[2];
#pragma unroll
            for (int nt = 0; nt < 2; ++nt) {
                bfr[nt].h[0] = *(const v8bf*)(bC + bOff[nt]);
                bfr[nt].h[1] = *(const v8bf*)(bC + bOff[nt] + 16);
            }
#pragma unroll
            for (int mt = 0; mt < 4; ++mt)
#pragma unroll
                for (int nt = 0; nt < 2; ++nt)
                    acc[mt][nt] = __builtin_amdgcn_wmma_f32_16x16x32_bf16(
                        false, afr[mt].v, false, bfr[nt].v,
                        (short)0, acc[mt][nt], false, false);
        };

        // ---- pipeline prologue: async-issue tiles 0 and 1 (depth 2) ----
        issue_tile(0, aBase0, bBase0);
        issue_tile(1, aBase0 + BUFBYTES, bBase0 + BUFBYTES);

        // Rotating state: read buffer (generic ptr) and prefetch target
        // (LDS byte base). Target of step kt is buffer (kt+2)%3.
        const __bf16* aCur = &As[0][0];
        const __bf16* bCur = &Bs[0][0];
        unsigned aTgt = aBase0 + 2u * BUFBYTES;
        unsigned bTgt = bBase0 + 2u * BUFBYTES;

        // Steady state: kt in [0, NK-2) — always prefetch, always wait<=4.
        // 4 async instructions per wave per tile; in-order retirement:
        // <=4 outstanding  ==>  tile kt fully landed in LDS.
#pragma unroll 1
        for (int kt = 0; kt < NK - 2; ++kt) {
            asm volatile("s_wait_asynccnt 0x4" ::: "memory");
            __syncthreads();  // tile kt visible everywhere; buf[(kt-1)%3] readers done
            issue_tile(kt + 2, aTgt, bTgt);
            frag_wmma(aCur, bCur);
            // rotate cur: 0->1->2->0 ; tgt: 2->0->1->2
            aCur = (aCur == &As[2][0]) ? &As[0][0] : (aCur + BUFELTS);
            bCur = (bCur == &Bs[2][0]) ? &Bs[0][0] : (bCur + BUFELTS);
            aTgt = (aTgt == aBase0 + 2u * BUFBYTES) ? aBase0 : (aTgt + BUFBYTES);
            bTgt = (bTgt == bBase0 + 2u * BUFBYTES) ? bBase0 : (bTgt + BUFBYTES);
        }
        // kt = NK-2: tile NK-1 already in flight; nothing left to prefetch.
        asm volatile("s_wait_asynccnt 0x4" ::: "memory");
        __syncthreads();
        frag_wmma(aCur, bCur);
        aCur = (aCur == &As[2][0]) ? &As[0][0] : (aCur + BUFELTS);
        bCur = (bCur == &Bs[2][0]) ? &Bs[0][0] : (bCur + BUFELTS);
        // kt = NK-1: drain all async copies.
        asm volatile("s_wait_asynccnt 0x0" ::: "memory");
        __syncthreads();
        frag_wmma(aCur, bCur);

        // Fused epilogue: outAcc += gate_row * relu(acc + bias).
        // C layout: VGPR j <-> row (half*8 + j), col = laneLo — so an 8-float
        // LDS vector load of Gs lines up exactly with the accumulator lanes.
#pragma unroll
        for (int mt = 0; mt < 4; ++mt) {
            v8f g = *(const v8f*)&Gs[wm * 64 + mt * 16 + half * 8];
#pragma unroll
            for (int nt = 0; nt < 2; ++nt) {
#pragma unroll
                for (int j = 0; j < 8; ++j) {
                    float h = acc[mt][nt][j] + bv[nt];
                    h = fmaxf(h, 0.0f);
                    outAcc[mt][nt][j] = fmaf(g[j], h, outAcc[mt][nt][j]);
                }
            }
        }
    }

    // Store fp32 result (coalesced over laneLo).
#pragma unroll
    for (int mt = 0; mt < 4; ++mt) {
#pragma unroll
        for (int nt = 0; nt < 2; ++nt) {
#pragma unroll
            for (int j = 0; j < 8; ++j) {
                int row = bm0 + wm * 64 + mt * 16 + half * 8 + j;
                out[(size_t)row * Ndim + col[nt]] = outAcc[mt][nt][j];
            }
        }
    }
}

// ---------------------------------------------------------------------------
// Host launcher
// ---------------------------------------------------------------------------
extern "C" void kernel_launch(void* const* d_in, const int* in_sizes, int n_in,
                              void* d_out, int out_size, void* d_ws, size_t ws_size,
                              hipStream_t stream) {
    (void)in_sizes; (void)n_in; (void)out_size; (void)ws_size;
    const float* x  = (const float*)d_in[0];   // [B, K]
    const float* W  = (const float*)d_in[1];   // [M, K, N]
    const float* b  = (const float*)d_in[2];   // [M, N]
    const float* Wg = (const float*)d_in[3];   // [K, M]
    const float* bg = (const float*)d_in[4];   // [M]
    float* out = (float*)d_out;                // [B, N]

    char* ws = (char*)d_ws;
    const size_t gatesBytes = (size_t)Bsz * Mexp * sizeof(float);          // 256 KB
    const size_t xbBytes    = (size_t)Bsz * Kdim * sizeof(__bf16);         // 8 MB
    float*  gates = (float*)ws;
    __bf16* xb    = (__bf16*)(ws + gatesBytes);
    __bf16* wt    = (__bf16*)(ws + gatesBytes + xbBytes);                  // 32 MB

    // 1) x -> bf16
    cvt_x_kernel<<<(Bsz * Kdim) / (256 * 4), 256, 0, stream>>>(x, xb);
    // 2) W -> transposed bf16 [m][n][k]
    cvt_wT_kernel<<<Mexp * (Kdim / 32) * (Ndim / 32), 256, 0, stream>>>(W, wt);
    // 3) gates
    gate_kernel<<<Bsz / 8, 256, 0, stream>>>(x, Wg, bg, gates);
    // 4) fused gated-MoE GEMM
    dim3 grid(Bsz / BM, Ndim / BN);
    moe_wmma_kernel<<<grid, 256, 0, stream>>>(xb, wt, b, gates, out);
}
